// DiffeqSolver_64587718197730
// MI455X (gfx1250) — matrix operand order repaired and checked by reading, not compiled, not run on hardware
//
#include <hip/hip_runtime.h>
#include <hip/hip_bf16.h>
#include <math.h>

// Problem constants (from the reference): B=64, X=1024, H=4096, T=64.
#define BB 64
#define XX 1024
#define HH 4096
#define TT 64

typedef __attribute__((ext_vector_type(16))) __bf16 v16bf;
typedef __attribute__((ext_vector_type(8)))  __bf16 bf16x8;
typedef __attribute__((ext_vector_type(8)))  float  v8f;

// Hardware tanh (V_TANH_F32 is a CDNA5 TRANS op that co-executes with the
// matrix pipe); fall back to libm expansion if the builtin is unavailable.
__device__ __forceinline__ float hw_tanh(float x) {
#if __has_builtin(__builtin_amdgcn_tanhf)
  return __builtin_amdgcn_tanhf(x);
#else
  return tanhf(x);
#endif
}

// ---------------------------------------------------------------------------
// Weight prep: fp32 row-major (R x C) -> bf16 transposed (C x R), LDS-tiled so
// both the global reads and writes are coalesced. Runs once per launch; the
// bf16 copies (16 MB total) then live in L2 for all 504 GEMMs.
// ---------------------------------------------------------------------------
__global__ __launch_bounds__(256) void transpose_cvt_kernel(
    const float* __restrict__ in, __bf16* __restrict__ outT, int R, int C) {
  __shared__ float tile[32][33];
  int tc = blockIdx.x * 32;  // column base in `in`
  int tr = blockIdx.y * 32;  // row base in `in`
  int lx = threadIdx.x;      // 0..31
  int ly = threadIdx.y;      // 0..7
#pragma unroll
  for (int i = 0; i < 32; i += 8)
    tile[ly + i][lx] = in[(size_t)(tr + ly + i) * C + (tc + lx)];
  __syncthreads();
#pragma unroll
  for (int i = 0; i < 32; i += 8)
    outT[(size_t)(tc + ly + i) * R + (tr + lx)] = (__bf16)tile[lx][ly + i];
}

// pred[:,0,:] = x ; A0 = bf16(x)
__global__ __launch_bounds__(256) void init_kernel(
    const float* __restrict__ x, float* __restrict__ out,
    __bf16* __restrict__ a0) {
  int i = blockIdx.x * 256 + threadIdx.x;  // over B*X
  int b = i >> 10;
  int n = i & (XX - 1);
  float v = x[i];
  out[((size_t)b * TT + 0) * XX + n] = v;
  a0[i] = (__bf16)v;
}

// ---------------------------------------------------------------------------
// G1: Hout(64x4096) = tanh( A(64x1024) @ W1 + b1 ), bf16 in, bf16 out.
// Block = 128 threads (4 wave32). Wave w owns M rows 16w..16w+15; block owns
// an N-tile of 64 -> 4 accumulators (v8f) per wave, K-loop of 32 per WMMA.
// ---------------------------------------------------------------------------
__global__ __launch_bounds__(128) void g1_kernel(
    const __bf16* __restrict__ A,     // B x X (row-major, bf16)
    const __bf16* __restrict__ W1t,   // H x X (N x K, row-major bf16)
    const float* __restrict__ bias,   // H
    __bf16* __restrict__ Hout) {      // B x H
  const int K = XX, N = HH;
  int wave = threadIdx.x >> 5;
  int lane = threadIdx.x & 31;
  int mBase = wave * 16;
  int nBase = blockIdx.x * 64;
  int l16 = lane & 15;
  int hi = lane >> 4;

  v8f acc[4] = {};
  // A fragment per ISA layout: lanes 0-15 hold K {0..7,16..23}, lanes 16-31
  // hold K {8..15,24..31} for their row M = lane&15.
  const __bf16* aRow = A + (size_t)(mBase + l16) * K + (hi ? 8 : 0);

  for (int kb = 0; kb < K; kb += 32) {
    v16bf afrag;
    ((bf16x8*)&afrag)[0] = *(const bf16x8*)(aRow + kb);
    ((bf16x8*)&afrag)[1] = *(const bf16x8*)(aRow + kb + 16);
#pragma unroll
    for (int s = 0; s < 4; ++s) {
      // B fragment: lane = column n, contiguous K run (thanks to transpose).
      const __bf16* bp = W1t + (size_t)(nBase + s * 16 + l16) * K + kb + hi * 8;
      v16bf bfrag;
      ((bf16x8*)&bfrag)[0] = *(const bf16x8*)(bp);
      ((bf16x8*)&bfrag)[1] = *(const bf16x8*)(bp + 16);
      acc[s] = __builtin_amdgcn_wmma_f32_16x16x32_bf16(
          false, afrag, false, bfrag, (short)0, acc[s], false, false);
    }
  }

#pragma unroll
  for (int s = 0; s < 4; ++s) {
    int n = nBase + s * 16 + l16;
    float bv = bias[n];
#pragma unroll
    for (int r = 0; r < 8; ++r) {
      int m = mBase + r + hi * 8;  // C/D layout: VGPR r -> M = r + 8*hi
      float v = acc[s][r] + bv;
      Hout[(size_t)m * N + n] = (__bf16)hw_tanh(v);
    }
  }
}

// ---------------------------------------------------------------------------
// G2: k(64x1024) = Hin(64x4096) @ W2 + b2, fused with the RK4 stage update:
//   stage 1: acc = dt/6*k;      a_next = y + dt/2*k
//   stage 2: acc += dt/3*k;     a_next = y + dt/2*k
//   stage 3: acc += dt/3*k;     a_next = y + dt*k
//   stage 4: y1 = y + acc + dt/6*k ; pred[:,t+1,:] = y1 ; a_next = bf16(y1)
// Block = 128 threads (4 waves); each wave does one 16x16 tile, K=4096.
// ---------------------------------------------------------------------------
__global__ __launch_bounds__(128) void g2_kernel(
    const __bf16* __restrict__ Hin,   // B x H bf16
    const __bf16* __restrict__ W2t,   // X x H (N x K) bf16
    const float* __restrict__ bias,   // X
    const float* __restrict__ ts,     // T
    int t, int stage,
    float* __restrict__ out,          // B x T x X (pred; y lives in slices)
    float* __restrict__ acc,          // B x X fp32 RK4 accumulator
    __bf16* __restrict__ Anext) {     // B x X bf16 (next stage A operand)
  const int K = HH, N = XX;
  int wave = threadIdx.x >> 5;
  int lane = threadIdx.x & 31;
  int mBase = wave * 16;
  int nTile = blockIdx.x * 16;
  int l16 = lane & 15;
  int hi = lane >> 4;

  v8f c = {};
  const __bf16* aRow = Hin + (size_t)(mBase + l16) * K + (hi ? 8 : 0);
  const __bf16* bRow = W2t + (size_t)(nTile + l16) * K + hi * 8;

  for (int kb = 0; kb < K; kb += 32) {
    v16bf af, bfm;
    ((bf16x8*)&af)[0] = *(const bf16x8*)(aRow + kb);
    ((bf16x8*)&af)[1] = *(const bf16x8*)(aRow + kb + 16);
    ((bf16x8*)&bfm)[0] = *(const bf16x8*)(bRow + kb);
    ((bf16x8*)&bfm)[1] = *(const bf16x8*)(bRow + kb + 16);
    c = __builtin_amdgcn_wmma_f32_16x16x32_bf16(
        false, af, false, bfm, (short)0, c, false, false);
  }

  float dt = ts[t + 1] - ts[t];
  int n = nTile + l16;
  float bv = bias[n];
  float wAcc = (stage == 1 || stage == 4) ? dt * (1.0f / 6.0f)
                                          : dt * (1.0f / 3.0f);
  float cIn = (stage == 3) ? dt : dt * 0.5f;

#pragma unroll
  for (int r = 0; r < 8; ++r) {
    int m = mBase + r + hi * 8;
    float kv = c[r] + bv;
    size_t idx = (size_t)m * N + n;
    float aPrev = (stage == 1) ? 0.0f : acc[idx];
    float aNew = aPrev + wAcc * kv;
    float ybase = out[((size_t)m * TT + t) * XX + n];
    float nxt;
    if (stage == 4) {
      nxt = ybase + aNew;                       // y_{t+1}
      out[((size_t)m * TT + t + 1) * XX + n] = nxt;
    } else {
      acc[idx] = aNew;
      nxt = ybase + cIn * kv;                   // next RK4 stage input
    }
    Anext[idx] = (__bf16)nxt;
  }
}

// ---------------------------------------------------------------------------
extern "C" void kernel_launch(void* const* d_in, const int* in_sizes, int n_in,
                              void* d_out, int out_size, void* d_ws,
                              size_t ws_size, hipStream_t stream) {
  (void)in_sizes; (void)n_in; (void)out_size; (void)ws_size;
  const float* x  = (const float*)d_in[0];
  const float* ts = (const float*)d_in[1];
  const float* W1 = (const float*)d_in[2];
  const float* b1 = (const float*)d_in[3];
  const float* W2 = (const float*)d_in[4];
  const float* b2 = (const float*)d_in[5];
  float* out = (float*)d_out;

  // Workspace layout (~17 MB):
  char* ws = (char*)d_ws;
  __bf16* W1t = (__bf16*)(ws);                          // H*X bf16 = 8 MB
  __bf16* W2t = (__bf16*)(ws + (size_t)8 * 1024 * 1024);// X*H bf16 = 8 MB
  __bf16* hbf = (__bf16*)(ws + (size_t)16 * 1024 * 1024);       // B*H bf16
  __bf16* abf = (__bf16*)(ws + (size_t)16 * 1024 * 1024 + 524288); // B*X bf16
  float*  acc = (float*)(ws + (size_t)16 * 1024 * 1024 + 524288 + 131072);

  // One-time (per launch) weight transpose+convert: W1t[n*X+k] = W1[k*H+n].
  transpose_cvt_kernel<<<dim3(HH / 32, XX / 32), dim3(32, 8), 0, stream>>>(
      W1, W1t, XX, HH);
  transpose_cvt_kernel<<<dim3(XX / 32, HH / 32), dim3(32, 8), 0, stream>>>(
      W2, W2t, HH, XX);
  init_kernel<<<(BB * XX) / 256, 256, 0, stream>>>(x, out, abf);

  for (int t = 0; t < TT - 1; ++t) {
    for (int stage = 1; stage <= 4; ++stage) {
      g1_kernel<<<HH / 64, 128, 0, stream>>>(abf, W1t, b1, hbf);
      g2_kernel<<<XX / 16, 128, 0, stream>>>(hbf, W2t, b2, ts, t, stage,
                                             out, acc, abf);
    }
  }
}
